// PSI_RandomPhase_35622458753035
// MI455X (gfx1250) — matrix operand order repaired and checked by reading, compiled, not run on hardware
//
#include <hip/hip_runtime.h>
#include <hip/hip_bf16.h>
#include <cstdint>

// ---------------------------------------------------------------------------
// Problem constants (reference: DIM=1024, B=4, S=4096)
// ---------------------------------------------------------------------------
#define DIMD 1024
#define BATCH 4
#define SEQ 4096
#define TOK (BATCH * SEQ)   // 16384 rows
#define SCAN_CH 512         // scan chunk length
#define SCAN_NC (SEQ / SCAN_CH)   // 8 chunks

typedef __attribute__((ext_vector_type(16))) __bf16 v16bf;
typedef __attribute__((ext_vector_type(8)))  float  v8f;

// ---------------------------------------------------------------------------
// bf16 helpers (round-to-nearest-even)
// ---------------------------------------------------------------------------
__device__ __forceinline__ unsigned short f2bf(float f) {
  union { float f; uint32_t u; } v; v.f = f;
  uint32_t u = v.u;
  uint32_t r = u + 0x7FFFu + ((u >> 16) & 1u);
  return (unsigned short)(r >> 16);
}
__device__ __forceinline__ float bf2f(unsigned short h) {
  union { uint32_t u; float f; } v; v.u = ((uint32_t)h) << 16;
  return v.f;
}

// async 16B/lane global->LDS copy (ASYNCcnt-tracked)
__device__ __forceinline__ void async_cp_b128(uint32_t lds_off, const void* g) {
  unsigned long long ga = (unsigned long long)(uintptr_t)g;
  asm volatile("global_load_async_to_lds_b128 %0, %1, off"
               :: "v"(lds_off), "v"(ga) : "memory");
}
__device__ __forceinline__ void wait_async_le(int n) {
#if __has_builtin(__builtin_amdgcn_s_wait_asynccnt)
  if (n == 0) __builtin_amdgcn_s_wait_asynccnt(0);
  else        __builtin_amdgcn_s_wait_asynccnt(4);
#else
  if (n == 0) asm volatile("s_wait_asynccnt 0" ::: "memory");
  else        asm volatile("s_wait_asynccnt 4" ::: "memory");
#endif
}

// ---------------------------------------------------------------------------
// fp32 -> bf16 conversion
// ---------------------------------------------------------------------------
__global__ __launch_bounds__(256) void cvt_f32_bf16_kernel(
    const float* __restrict__ src, unsigned short* __restrict__ dst, int n) {
  int i = blockIdx.x * 256 + threadIdx.x;
  if (i < n) dst[i] = f2bf(src[i]);
}

// ---------------------------------------------------------------------------
// WMMA bf16 GEMM:  C[M,N] = act(A[M,K] * W[N,K]^T + bias[N]) (+residual)
//   MODE 0: sigmoid(v)*5   -> fp32   (magnitude)
//   MODE 1: identity       -> fp32   (q_off)
//   MODE 2: exact GELU     -> bf16   (h1)
//   MODE 3: + residual     -> fp32   (final output)
// Block tile 128x128, 256 threads (8 waves). Wave w owns a 32x64 region:
// rows (w>>1)*32..+31, cols (w&1)*64..+63  => 2x4 grid of 16x16 WMMA tiles.
// K-step 32, double-buffered LDS filled with async b128 copies.
// ---------------------------------------------------------------------------
template <int MODE>
__global__ __launch_bounds__(256) void gemm_wmma_kernel(
    const unsigned short* __restrict__ A,    // [M,K] bf16
    const unsigned short* __restrict__ W,    // [N,K] bf16
    const float* __restrict__ bias,          // [N]
    const float* __restrict__ resid,         // [M,N] (MODE 3) or nullptr
    float* __restrict__ outF,                // fp32 out (MODE 0,1,3)
    unsigned short* __restrict__ outB,       // bf16 out (MODE 2)
    int M, int N, int K) {
  __shared__ __align__(16) unsigned short sA[2][128 * 32];
  __shared__ __align__(16) unsigned short sB[2][128 * 32];

  const int m0   = blockIdx.y * 128;
  const int n0   = blockIdx.x * 128;
  const int t    = threadIdx.x;
  const int lane = t & 31;
  const int lh   = lane >> 4;       // half-wave
  const int l15  = lane & 15;
  const int wave = t >> 5;          // 0..7
  const int wm   = (wave >> 1) * 32;   // 0,32,64,96
  const int wn   = (wave & 1) * 64;    // 0,64

  // cooperative copy mapping: thread t moves chunks {t, t+256} of each tile
  // chunk c: row = c>>2 (0..127), k-offset = (c&3)*8 elements (16B)
  const int cr  = t >> 2;          // 0..63 ; second chunk row = cr+64
  const int cke = (t & 3) * 8;

  uint32_t ldsA0[2], ldsA1[2], ldsB0[2], ldsB1[2];
#pragma unroll
  for (int bf = 0; bf < 2; ++bf) {
    ldsA0[bf] = (uint32_t)(uintptr_t)(&sA[bf][cr * 32 + cke]);
    ldsA1[bf] = (uint32_t)(uintptr_t)(&sA[bf][(cr + 64) * 32 + cke]);
    ldsB0[bf] = (uint32_t)(uintptr_t)(&sB[bf][cr * 32 + cke]);
    ldsB1[bf] = (uint32_t)(uintptr_t)(&sB[bf][(cr + 64) * 32 + cke]);
  }

  v8f acc[2][4];
#pragma unroll
  for (int ti = 0; ti < 2; ++ti)
#pragma unroll
    for (int tj = 0; tj < 4; ++tj) acc[ti][tj] = (v8f){};

  const int niter = K >> 5;        // K/32 (K is a multiple of 32)

  auto issue = [&](int it, int bf) {
    const int kb = it * 32;
    const unsigned short* gA0 = A + (size_t)(m0 + cr) * K + kb + cke;
    const unsigned short* gA1 = A + (size_t)(m0 + cr + 64) * K + kb + cke;
    const unsigned short* gB0 = W + (size_t)(n0 + cr) * K + kb + cke;
    const unsigned short* gB1 = W + (size_t)(n0 + cr + 64) * K + kb + cke;
    async_cp_b128(ldsA0[bf], gA0);
    async_cp_b128(ldsA1[bf], gA1);
    async_cp_b128(ldsB0[bf], gB0);
    async_cp_b128(ldsB1[bf], gB1);
    __builtin_prefetch(gA0 + 64, 0, 1);    // next-next K tile toward L2
    __builtin_prefetch(gB0 + 64, 0, 1);
  };

  auto compute = [&](int bf) {
    const uint32_t* a32 = (const uint32_t*)sA[bf];
    const uint32_t* b32 = (const uint32_t*)sB[bf];
    union { v16bf v; uint32_t u[8]; } af[2], bfragm[4];
    // A fragment (16x32): lanes 0-15 rows; VGPR j: K=2j+16*lh (lanes 0-15 K=0..15, lanes 16-31 K=16..31)
#pragma unroll
    for (int ti = 0; ti < 2; ++ti) {
      const int am = wm + ti * 16 + l15;
#pragma unroll
      for (int j = 0; j < 4; ++j) {
        af[ti].u[j]     = a32[am * 16 + j + 8 * lh];
        af[ti].u[4 + j] = a32[am * 16 + 4 + j + 8 * lh];
      }
    }
    // B fragment (32x16): lanes 0-15 hold K=0..15, lanes 16-31 K=16..31
#pragma unroll
    for (int tj = 0; tj < 4; ++tj) {
      const int bn = wn + tj * 16 + l15;
#pragma unroll
      for (int j = 0; j < 8; ++j) bfragm[tj].u[j] = b32[bn * 16 + j + 8 * lh];
    }
#pragma unroll
    for (int ti = 0; ti < 2; ++ti)
#pragma unroll
      for (int tj = 0; tj < 4; ++tj)
        acc[ti][tj] = __builtin_amdgcn_wmma_f32_16x16x32_bf16(
            false, af[ti].v, false, bfragm[tj].v, (short)0, acc[ti][tj],
            false, false);
  };

  issue(0, 0);
  for (int i = 0; i < niter - 1; ++i) {
    const int cur = i & 1;
    issue(i + 1, cur ^ 1);
    wait_async_le(4);    // in-order completion => current buffer landed
    __syncthreads();
    compute(cur);
    __syncthreads();     // everyone done reading before buffer is refilled
  }
  wait_async_le(0);
  __syncthreads();
  compute((niter - 1) & 1);

  // Epilogue. C/D layout: lane l, VGPR r -> C[r + 8*(l>>4)][l&15]
#pragma unroll
  for (int tj = 0; tj < 4; ++tj) {
    const int oc = n0 + wn + tj * 16 + l15;
    const float bv = bias[oc];
#pragma unroll
    for (int ti = 0; ti < 2; ++ti) {
#pragma unroll
      for (int r = 0; r < 8; ++r) {
        const int mr = m0 + wm + ti * 16 + r + 8 * lh;
        float f = acc[ti][tj][r] + bv;
        if (MODE == 0) {
          f = 5.0f / (1.0f + __expf(-f));
        } else if (MODE == 2) {
          f = 0.5f * f * (1.0f + erff(f * 0.70710678118654752f));
        } else if (MODE == 3) {
          f += resid[(size_t)mr * N + oc];
        }
        if (MODE == 2) outB[(size_t)mr * N + oc] = f2bf(f);
        else           outF[(size_t)mr * N + oc] = f;
      }
    }
  }
}

// ---------------------------------------------------------------------------
// Chunked scan, phase 1: per-(b,chunk,d) partial sums of weighted*cos/sin
// and magnitude over a chunk of SCAN_CH positions.
// ---------------------------------------------------------------------------
__global__ __launch_bounds__(256) void scan_partial_kernel(
    const float* __restrict__ x, const float* __restrict__ mag,
    const float* __restrict__ phases,
    float* __restrict__ pr, float* __restrict__ pi, float* __restrict__ pm) {
  const int idx = blockIdx.x * 256 + threadIdx.x;   // 0..B*NC*D-1
  const int d = idx % DIMD;
  const int c = (idx / DIMD) % SCAN_NC;
  const int b = idx / (DIMD * SCAN_NC);
  const size_t base = (size_t)b * SEQ * DIMD + d;
  const float step = phases[d] * 0.01f;
  float sr = 0.f, si = 0.f, sm = 0.f;
  const int s0 = c * SCAN_CH;
  for (int s = s0; s < s0 + SCAN_CH; ++s) {
    const size_t off = base + (size_t)s * DIMD;
    const float xv = x[off];
    const float m  = mag[off];
    const float ph = step * (float)s;
    const float w  = m * xv;
    sr += w * __cosf(ph);
    si += w * __sinf(ph);
    sm += m;
  }
  pr[idx] = sr; pi[idx] = si; pm[idx] = sm;
}

// phase 2: exclusive prefix over the SCAN_NC chunks, in place
__global__ __launch_bounds__(256) void scan_prefix_kernel(
    float* __restrict__ pr, float* __restrict__ pi, float* __restrict__ pm) {
  const int idx = blockIdx.x * 256 + threadIdx.x;   // 0..B*D-1
  const int d = idx % DIMD;
  const int b = idx / DIMD;
  float ar = 0.f, ai = 0.f, am = 0.f;
#pragma unroll
  for (int c = 0; c < SCAN_NC; ++c) {
    const size_t o = ((size_t)b * SCAN_NC + c) * DIMD + d;
    const float tr = pr[o], ti = pi[o], tm = pm[o];
    pr[o] = ar; pi[o] = ai; pm[o] = am;
    ar += tr; ai += ti; am += tm;
  }
}

// phase 3: re-scan each chunk starting from its exclusive offsets and emit
// the 4D-wide context row pieces [x*cos, x*sin, retrieved_r, retrieved_i] bf16
__global__ __launch_bounds__(256) void scan_emit_kernel(
    const float* __restrict__ x, const float* __restrict__ mag,
    const float* __restrict__ qoff, const float* __restrict__ phases,
    const float* __restrict__ pr, const float* __restrict__ pi,
    const float* __restrict__ pm, unsigned short* __restrict__ ctx) {
  const int idx = blockIdx.x * 256 + threadIdx.x;   // 0..B*NC*D-1
  const int d = idx % DIMD;
  const int c = (idx / DIMD) % SCAN_NC;
  const int b = idx / (DIMD * SCAN_NC);
  const size_t base = (size_t)b * SEQ * DIMD + d;
  const float step = phases[d] * 0.01f;
  float mem_r = pr[idx], mem_i = pi[idx], acc_mag = pm[idx];
  const int s0 = c * SCAN_CH;
  for (int s = s0; s < s0 + SCAN_CH; ++s) {
    const size_t off = base + (size_t)s * DIMD;
    const float xv = x[off];
    const float m  = mag[off];
    const float q  = qoff[off];
    const float ph = step * (float)s;
    const float cp = __cosf(ph);
    const float sp = __sinf(ph);
    const float w  = m * xv;
    mem_r   += w * cp;
    mem_i   += w * sp;
    acc_mag += m;
    const float inv = rsqrtf(acc_mag + 1e-8f);
    const float mr = mem_r * inv;
    const float mi = mem_i * inv;
    const float cq = __cosf(ph + q);
    const float sq = __sinf(ph + q);
    const float rr = mr * cq + mi * sq;
    const float ri = mi * cq - mr * sq;
    unsigned short* row = ctx + (size_t)(b * SEQ + s) * (4 * DIMD);
    row[d]            = f2bf(xv * cp);
    row[DIMD + d]     = f2bf(xv * sp);
    row[2 * DIMD + d] = f2bf(rr);
    row[3 * DIMD + d] = f2bf(ri);
  }
}

// ---------------------------------------------------------------------------
// LayerNorm over F=4096 features, in-place on bf16 buffer. One block per row.
// ---------------------------------------------------------------------------
__global__ __launch_bounds__(256) void ln_kernel(
    unsigned short* __restrict__ ctx,
    const float* __restrict__ g, const float* __restrict__ bb) {
  const int F = 4 * DIMD;                       // 4096
  unsigned short* p = ctx + (size_t)blockIdx.x * F;
  float vals[16];
  float s = 0.f, s2 = 0.f;
#pragma unroll
  for (int i = 0; i < 16; ++i) {
    const float v = bf2f(p[threadIdx.x + i * 256]);
    vals[i] = v; s += v; s2 += v * v;
  }
#pragma unroll
  for (int o = 16; o > 0; o >>= 1) {   // wave32 reduction
    s  += __shfl_down(s, o);
    s2 += __shfl_down(s2, o);
  }
  __shared__ float rs[8], rs2[8];
  if ((threadIdx.x & 31) == 0) {
    rs[threadIdx.x >> 5]  = s;
    rs2[threadIdx.x >> 5] = s2;
  }
  __syncthreads();
  float ts = 0.f, ts2 = 0.f;
#pragma unroll
  for (int i = 0; i < 8; ++i) { ts += rs[i]; ts2 += rs2[i]; }
  const float mu  = ts * (1.0f / (float)F);
  const float var = ts2 * (1.0f / (float)F) - mu * mu;
  const float inv = rsqrtf(var + 1e-5f);
#pragma unroll
  for (int i = 0; i < 16; ++i) {
    const int j = threadIdx.x + i * 256;
    const float h = (vals[i] - mu) * inv * g[j] + bb[j];
    p[j] = f2bf(h);
  }
}

// ---------------------------------------------------------------------------
// Host-side launcher
// ---------------------------------------------------------------------------
extern "C" void kernel_launch(void* const* d_in, const int* in_sizes, int n_in,
                              void* d_out, int out_size, void* d_ws, size_t ws_size,
                              hipStream_t stream) {
  (void)in_sizes; (void)n_in; (void)out_size; (void)ws_size;
  const int D = DIMD, T = TOK;

  const float* x      = (const float*)d_in[0];
  const float* phases = (const float*)d_in[1];
  const float* W_mag  = (const float*)d_in[2];
  const float* b_mag  = (const float*)d_in[3];
  const float* W_q    = (const float*)d_in[4];
  const float* b_q    = (const float*)d_in[5];
  const float* ln_g   = (const float*)d_in[6];
  const float* ln_b   = (const float*)d_in[7];
  const float* W1     = (const float*)d_in[8];
  const float* b1     = (const float*)d_in[9];
  const float* W2     = (const float*)d_in[10];
  const float* b2     = (const float*)d_in[11];
  float* out          = (float*)d_out;

  // workspace carve-up (256B aligned)
  char* w = (char*)d_ws;
  size_t off = 0;
  auto alloc = [&](size_t bytes) {
    void* p = w + off;
    off = (off + bytes + 255) & ~(size_t)255;
    return p;
  };
  unsigned short* xb   = (unsigned short*)alloc((size_t)T * D * 2);
  unsigned short* wmb  = (unsigned short*)alloc((size_t)D * D * 2);
  unsigned short* wqb  = (unsigned short*)alloc((size_t)D * D * 2);
  unsigned short* w1b  = (unsigned short*)alloc((size_t)2 * D * 4 * D * 2);
  unsigned short* w2b  = (unsigned short*)alloc((size_t)D * 2 * D * 2);
  float*          magf = (float*)alloc((size_t)T * D * 4);
  float*          qf   = (float*)alloc((size_t)T * D * 4);
  unsigned short* ctx  = (unsigned short*)alloc((size_t)T * 4 * D * 2);
  unsigned short* h1   = (unsigned short*)alloc((size_t)T * 2 * D * 2);
  float* pr = (float*)alloc((size_t)BATCH * SCAN_NC * D * 4);
  float* pi = (float*)alloc((size_t)BATCH * SCAN_NC * D * 4);
  float* pm = (float*)alloc((size_t)BATCH * SCAN_NC * D * 4);

  auto cvt = [&](const float* s, unsigned short* dptr, int n) {
    cvt_f32_bf16_kernel<<<(n + 255) / 256, 256, 0, stream>>>(s, dptr, n);
  };
  cvt(x,     xb,  T * D);
  cvt(W_mag, wmb, D * D);
  cvt(W_q,   wqb, D * D);
  cvt(W1,    w1b, 2 * D * 4 * D);
  cvt(W2,    w2b, D * 2 * D);

  // magnitude = sigmoid(x W_mag^T + b)*5 ;  q_off = x W_q^T + b
  gemm_wmma_kernel<0><<<dim3(D / 128, T / 128), 256, 0, stream>>>(
      xb, wmb, b_mag, nullptr, magf, nullptr, T, D, D);
  gemm_wmma_kernel<1><<<dim3(D / 128, T / 128), 256, 0, stream>>>(
      xb, wqb, b_q, nullptr, qf, nullptr, T, D, D);

  // chunked phase-memory scan -> context [T, 4D] bf16
  const int lanes = BATCH * SCAN_NC * D;   // 32768
  scan_partial_kernel<<<lanes / 256, 256, 0, stream>>>(x, magf, phases, pr, pi, pm);
  scan_prefix_kernel<<<(BATCH * D) / 256, 256, 0, stream>>>(pr, pi, pm);
  scan_emit_kernel<<<lanes / 256, 256, 0, stream>>>(x, magf, qf, phases,
                                                    pr, pi, pm, ctx);

  // LayerNorm in place
  ln_kernel<<<T, 256, 0, stream>>>(ctx, ln_g, ln_b);

  // h1 = GELU(ctx W1^T + b1)  -> bf16 [T, 2D]
  gemm_wmma_kernel<2><<<dim3(2 * D / 128, T / 128), 256, 0, stream>>>(
      ctx, w1b, b1, nullptr, nullptr, h1, T, 2 * D, 4 * D);

  // out = x + h1 W2^T + b2  -> fp32 [T, D]
  gemm_wmma_kernel<3><<<dim3(D / 128, T / 128), 256, 0, stream>>>(
      h1, w2b, b2, x, out, nullptr, T, D, 2 * D);
}
